// CAM_GATSM_47115791237653
// MI455X (gfx1250) — matrix operand order repaired and checked by reading, not compiled, not run on hardware
//
#include <hip/hip_runtime.h>
#include <math.h>

// ---------------- problem constants ----------------
static constexpr int B_  = 32;
static constexpr int T_  = 512;
static constexpr int D_  = 1024;
static constexpr int P_  = 512;
static constexpr int E_  = 256;
static constexpr int K_  = 14;
static constexpr int NH_ = 4;
static constexpr int L_  = 2;
static constexpr int FF_ = 2048;
static constexpr int TOPK_ = 7;
static constexpr int FOLD_ = D_ / 2;          // 512
static constexpr int BT_ = B_ * T_;           // 16384
static constexpr float EPS_ = 1e-5f;

// ---------------- WMMA types ----------------
typedef __attribute__((ext_vector_type(16))) __bf16          v16bf;
typedef __attribute__((ext_vector_type(16))) unsigned short  v16u;
typedef __attribute__((ext_vector_type(8)))  float           v8f;
typedef __attribute__((ext_vector_type(4)))  unsigned int    v4u;

union Frag { v4u q[2]; v16u u; v16bf b; };

__device__ __forceinline__ unsigned short f2bf(float f) {
  unsigned int u = __float_as_uint(f);
  unsigned int r = u + 0x7FFFu + ((u >> 16) & 1u);   // round-to-nearest-even
  return (unsigned short)(r >> 16);
}
__device__ __forceinline__ unsigned int pack2bf(float f0, float f1) {
  return (unsigned int)f2bf(f0) | ((unsigned int)f2bf(f1) << 16);
}

// =====================================================================
// Generic bf16 WMMA GEMM:  C = act(alpha * A@B + bias)
// block = 128 threads (4 waves), block tile 128(M) x 64(N), K-step 32.
// wave tile 64x32 -> 4x2 accumulators of v_wmma_f32_16x16x32_bf16.
// A tile  : LDS [row][k]   pitch 40 ushorts (80B, 16B aligned)
// B tile  : LDS [n][k] (transposed) pitch 40 ushorts
// -> every fragment is 2x contiguous 128-bit LDS loads (no repacking).
// batched via blockIdx.z = zb*nh + zh with separate batch/head strides.
// =====================================================================
__global__ __launch_bounds__(128)
void gemm_wmma_kernel(const float* __restrict__ A, long long lda, long long sAb, long long sAh,
                      const float* __restrict__ Bm, long long ldb, long long sBb, long long sBh,
                      int transB,
                      const float* __restrict__ bias,
                      float* __restrict__ C, long long ldc, long long sCb, long long sCh,
                      int M, int N, int K, int nh, float alpha, int act) {
  __shared__ __align__(16) unsigned short As[128][40];
  __shared__ __align__(16) unsigned short Bst[64][40];

  const int z  = blockIdx.z;
  const int zb = z / nh;
  const int zh = z - zb * nh;
  A  += (long long)zb * sAb + (long long)zh * sAh;
  Bm += (long long)zb * sBb + (long long)zh * sBh;
  C  += (long long)zb * sCb + (long long)zh * sCh;

  const int bM = blockIdx.y * 128;
  const int bN = blockIdx.x * 64;

  const int tid  = threadIdx.x;
  const int w    = tid >> 5;          // wave id 0..3
  const int lane = tid & 31;
  const int wm   = w >> 1;            // 0..1 (covers 64 rows)
  const int wn   = w & 1;             // 0..1 (covers 32 cols)
  const int half = lane >> 4;         // lane group 0/1
  const int lidx = lane & 15;

  v8f acc[4][2];
#pragma unroll
  for (int i = 0; i < 4; ++i)
#pragma unroll
    for (int j = 0; j < 2; ++j)
#pragma unroll
      for (int e = 0; e < 8; ++e) acc[i][j][e] = 0.0f;

  for (int k0 = 0; k0 < K; k0 += 32) {
    // ---- stage A tile (128 x 32): one row per thread, packed b128 stores ----
    {
      const int gr = bM + tid;
      const float* ap = A + (long long)gr * lda + k0;
      unsigned int pk[16];
      if (gr < M && (k0 + 32) <= K) {
        if (k0 + 32 < K) __builtin_prefetch(ap + 32, 0, 3);
#pragma unroll
        for (int j = 0; j < 16; ++j) pk[j] = pack2bf(ap[2 * j], ap[2 * j + 1]);
      } else {
#pragma unroll
        for (int j = 0; j < 16; ++j) {
          const float f0 = (gr < M && (k0 + 2 * j)     < K) ? ap[2 * j]     : 0.0f;
          const float f1 = (gr < M && (k0 + 2 * j + 1) < K) ? ap[2 * j + 1] : 0.0f;
          pk[j] = pack2bf(f0, f1);
        }
      }
      v4u* dst = (v4u*)&As[tid][0];
#pragma unroll
      for (int q = 0; q < 4; ++q) {
        v4u t;
        t[0] = pk[4 * q]; t[1] = pk[4 * q + 1]; t[2] = pk[4 * q + 2]; t[3] = pk[4 * q + 3];
        dst[q] = t;
      }
    }
    // ---- stage B tile (32 x 64) transposed: thread -> (column, k-half) ----
    {
      const int c  = tid >> 1;
      const int kh = (tid & 1) << 4;           // 0 or 16
      const int gn = bN + c;
      unsigned int pk[8];
      if (gn < N && (k0 + kh + 16) <= K) {
        if (transB) {
          const float* bp = Bm + (long long)gn * ldb + (k0 + kh);
#pragma unroll
          for (int j = 0; j < 8; ++j) pk[j] = pack2bf(bp[2 * j], bp[2 * j + 1]);
        } else {
          const float* bp = Bm + (long long)(k0 + kh) * ldb + gn;
#pragma unroll
          for (int j = 0; j < 8; ++j)
            pk[j] = pack2bf(bp[(long long)(2 * j) * ldb], bp[(long long)(2 * j + 1) * ldb]);
        }
      } else {
#pragma unroll
        for (int j = 0; j < 8; ++j) {
          const int gk0 = k0 + kh + 2 * j;
          const int gk1 = gk0 + 1;
          float f0 = 0.0f, f1 = 0.0f;
          if (gn < N && gk0 < K)
            f0 = transB ? Bm[(long long)gn * ldb + gk0] : Bm[(long long)gk0 * ldb + gn];
          if (gn < N && gk1 < K)
            f1 = transB ? Bm[(long long)gn * ldb + gk1] : Bm[(long long)gk1 * ldb + gn];
          pk[j] = pack2bf(f0, f1);
        }
      }
      v4u* dst = (v4u*)&Bst[c][kh];
#pragma unroll
      for (int q = 0; q < 2; ++q) {
        v4u t;
        t[0] = pk[4 * q]; t[1] = pk[4 * q + 1]; t[2] = pk[4 * q + 2]; t[3] = pk[4 * q + 3];
        dst[q] = t;
      }
    }
    __syncthreads();

    // ---- fragment loads: 2x ds_load_b128 each (CDNA5 wave32 layouts) ----
    // A: lane m = lidx, elements e<8 -> k=8*half+e ; e>=8 -> k=16+8*half+(e-8)
    Frag a[4];
#pragma unroll
    for (int i = 0; i < 4; ++i) {
      const int row = wm * 64 + i * 16 + lidx;
      const char* base = (const char*)&As[0][0] + row * 80;
      a[i].q[0] = *(const v4u*)(base + 16 * half);
      a[i].q[1] = *(const v4u*)(base + 32 + 16 * half);
    }
    // B: lane n = lidx, elements e<8 -> k=8*half+e ; e>=8 -> k=16+8*half+(e-8)
    Frag b[2];
#pragma unroll
    for (int j = 0; j < 2; ++j) {
      const int col = wn * 32 + j * 16 + lidx;
      const char* base = (const char*)&Bst[0][0] + col * 80;
      b[j].q[0] = *(const v4u*)(base + 16 * half);
      b[j].q[1] = *(const v4u*)(base + 32 + 16 * half);
    }

    // ---- 8 WMMAs per wave per K-step ----
#pragma unroll
    for (int i = 0; i < 4; ++i)
#pragma unroll
      for (int j = 0; j < 2; ++j)
        acc[i][j] = __builtin_amdgcn_wmma_f32_16x16x32_bf16(
            false, a[i].b, false, b[j].b, (short)0, acc[i][j], false, false);

    __syncthreads();
  }

  // ---- epilogue: alpha, bias, relu, bounds-checked store ----
#pragma unroll
  for (int j = 0; j < 2; ++j) {
    const int n = bN + wn * 32 + j * 16 + lidx;
    const float bv = (bias != nullptr && n < N) ? bias[n] : 0.0f;
#pragma unroll
    for (int i = 0; i < 4; ++i)
#pragma unroll
      for (int jv = 0; jv < 8; ++jv) {
        const int m = bM + wm * 64 + i * 16 + jv + (half << 3);
        if (m < M && n < N) {
          float v = alpha * acc[i][j][jv] + bv;
          if (act == 1) v = fmaxf(v, 0.0f);
          C[(long long)m * ldc + n] = v;
        }
      }
  }
}

// =====================================================================
// block reductions (blockDim.x == 256)
// =====================================================================
__device__ __forceinline__ float blk_sum(float v, float* red) {
  const int tid = threadIdx.x;
  red[tid] = v; __syncthreads();
  for (int s = 128; s > 0; s >>= 1) {
    if (tid < s) red[tid] += red[tid + s];
    __syncthreads();
  }
  float r = red[0]; __syncthreads();
  return r;
}
__device__ __forceinline__ float blk_max(float v, float* red) {
  const int tid = threadIdx.x;
  red[tid] = v; __syncthreads();
  for (int s = 128; s > 0; s >>= 1) {
    if (tid < s) red[tid] = fmaxf(red[tid], red[tid + s]);
    __syncthreads();
  }
  float r = red[0]; __syncthreads();
  return r;
}

// ------------- GlobalAdaptiveTSM: per-token adaptive channel shift -------------
__global__ __launch_bounds__(256)
void tsm_kernel(const float* __restrict__ x, const float* __restrict__ pw,
                const float* __restrict__ pb, float* __restrict__ h) {
  __shared__ float red[256];
  __shared__ int s_sn;
  const int row = blockIdx.x;                 // b*T + t
  const int t = row & (T_ - 1);
  const float* xr = x + (long long)row * D_;
  float s = 0.0f;
  for (int c = threadIdx.x; c < D_; c += 256) s += xr[c] * pw[c];
  const float dot = blk_sum(s, red);
  if (threadIdx.x == 0) {
    const float ratio = 1.0f / (1.0f + __expf(-(dot + pb[0])));
    s_sn = (int)(ratio * (float)FOLD_);
  }
  __syncthreads();
  const int sn = s_sn;
  const float* xp = xr - D_;  // valid iff t > 0
  const float* xn = xr + D_;  // valid iff t < T-1
  for (int c = threadIdx.x; c < D_; c += 256) {
    float v;
    if (c < sn)            v = (t > 0)      ? xp[c] : 0.0f;
    else if (c < 2 * sn)   v = (t < T_ - 1) ? xn[c] : 0.0f;
    else                   v = xr[c];
    h[(long long)row * D_ + c] = v;
  }
}

// ------------- LayerNorm(h + a) in place on h, row length D_ -------------
__global__ __launch_bounds__(256)
void ln_res_kernel(float* __restrict__ h, const float* __restrict__ a,
                   const float* __restrict__ g, const float* __restrict__ bta) {
  __shared__ float red[256];
  const long long base = (long long)blockIdx.x * D_;
  float vals[4];
  float s = 0.0f;
#pragma unroll
  for (int k = 0; k < 4; ++k) {
    const int c = threadIdx.x + k * 256;
    vals[k] = h[base + c] + a[base + c];
    s += vals[k];
  }
  const float mean = blk_sum(s, red) * (1.0f / (float)D_);
  float vs = 0.0f;
#pragma unroll
  for (int k = 0; k < 4; ++k) { const float d = vals[k] - mean; vs += d * d; }
  const float var = blk_sum(vs, red) * (1.0f / (float)D_);
  const float inv = rsqrtf(var + EPS_);
#pragma unroll
  for (int k = 0; k < 4; ++k) {
    const int c = threadIdx.x + k * 256;
    h[base + c] = (vals[k] - mean) * inv * g[c] + bta[c];
  }
}

// ------------- row softmax in place, one block per row -------------
__global__ __launch_bounds__(256)
void softmax_rows_kernel(float* __restrict__ x, int cols) {
  __shared__ float red[256];
  float* r = x + (long long)blockIdx.x * cols;
  float m = -3.4e38f;
  for (int c = threadIdx.x; c < cols; c += 256) m = fmaxf(m, r[c]);
  m = blk_max(m, red);
  float s = 0.0f;
  for (int c = threadIdx.x; c < cols; c += 256) {
    const float e = __expf(r[c] - m);
    r[c] = e; s += e;
  }
  s = blk_sum(s, red);
  const float inv = 1.0f / s;
  for (int c = threadIdx.x; c < cols; c += 256) r[c] *= inv;
}

// ------------- eval BatchNorm + ReLU (per-channel affine), in place -------------
__global__ __launch_bounds__(256)
void bn_relu_kernel(float* __restrict__ x, const float* __restrict__ g,
                    const float* __restrict__ b, const float* __restrict__ m,
                    const float* __restrict__ v, long long n, int C) {
  const long long i = (long long)blockIdx.x * blockDim.x + threadIdx.x;
  if (i >= n) return;
  const int c = (int)(i % C);
  const float sc = g[c] * rsqrtf(v[c] + EPS_);
  const float sh = b[c] - m[c] * sc;
  x[i] = fmaxf(x[i] * sc + sh, 0.0f);
}

// ------------- out[row] = x[row,:].w + bias -------------
__global__ __launch_bounds__(256)
void rowdot_kernel(const float* __restrict__ x, const float* __restrict__ w,
                   const float* __restrict__ bias, float* __restrict__ out, int cols) {
  __shared__ float red[256];
  const float* r = x + (long long)blockIdx.x * cols;
  float s = 0.0f;
  for (int c = threadIdx.x; c < cols; c += 256) s += r[c] * w[c];
  s = blk_sum(s, red);
  if (threadIdx.x == 0) out[blockIdx.x] = s + bias[0];
}

// ------------- top-7 over T, mean-pool shared rows -> agg[B,P] -------------
__global__ __launch_bounds__(256)
void topk_agg_kernel(const float* __restrict__ shr, const float* __restrict__ scores,
                     float* __restrict__ agg) {
  __shared__ float sc[T_];
  __shared__ float rv[256];
  __shared__ int   ri[256];
  __shared__ int   sel[TOPK_];
  const int b = blockIdx.x;
  for (int c = threadIdx.x; c < T_; c += 256) sc[c] = scores[b * T_ + c];
  __syncthreads();
  for (int it = 0; it < TOPK_; ++it) {
    float best = -3.4e38f; int bi = 0;
    for (int c = threadIdx.x; c < T_; c += 256)
      if (sc[c] > best) { best = sc[c]; bi = c; }
    rv[threadIdx.x] = best; ri[threadIdx.x] = bi;
    __syncthreads();
    for (int s = 128; s > 0; s >>= 1) {
      if (threadIdx.x < s && rv[threadIdx.x + s] > rv[threadIdx.x]) {
        rv[threadIdx.x] = rv[threadIdx.x + s];
        ri[threadIdx.x] = ri[threadIdx.x + s];
      }
      __syncthreads();
    }
    if (threadIdx.x == 0) { sel[it] = ri[0]; sc[ri[0]] = -3.4e38f; }
    __syncthreads();
  }
  for (int p = threadIdx.x; p < P_; p += 256) {
    float s = 0.0f;
    for (int k = 0; k < TOPK_; ++k)
      s += shr[((long long)b * T_ + sel[k]) * P_ + p];
    agg[b * P_ + p] = s * (1.0f / (float)TOPK_);
  }
}

// ------------- cat_emb = softmax(logits) @ ce -------------
__global__ __launch_bounds__(256)
void catemb_kernel(const float* __restrict__ logits, const float* __restrict__ ce,
                   float* __restrict__ cat) {
  __shared__ float p[K_];
  const int b = blockIdx.x;
  if (threadIdx.x == 0) {
    float m = -3.4e38f;
    for (int k = 0; k < K_; ++k) m = fmaxf(m, logits[b * K_ + k]);
    float s = 0.0f;
    for (int k = 0; k < K_; ++k) { p[k] = __expf(logits[b * K_ + k] - m); s += p[k]; }
    const float inv = 1.0f / s;
    for (int k = 0; k < K_; ++k) p[k] *= inv;
  }
  __syncthreads();
  for (int e = threadIdx.x; e < E_; e += 256) {
    float s = 0.0f;
    for (int k = 0; k < K_; ++k) s += p[k] * ce[k * E_ + e];
    cat[b * E_ + e] = s;
  }
}

// ------------- cross-attention scores: s[b,h,t] = q[b,h]·k[b,t,h] / sqrt(64) -------------
__global__ __launch_bounds__(256)
void cross_scores_kernel(const float* __restrict__ q, const float* __restrict__ kb,
                         float* __restrict__ s) {
  const int idx = blockIdx.x * blockDim.x + threadIdx.x;   // B*NH*T
  if (idx >= B_ * NH_ * T_) return;
  const int b = idx / (NH_ * T_);
  const int r = idx - b * NH_ * T_;
  const int hh = r / T_;
  const int t  = r - hh * T_;
  const int hd = E_ / NH_;  // 64
  float acc = 0.0f;
  const float* qp = q + b * E_ + hh * hd;
  const float* kp = kb + ((long long)b * T_ + t) * E_ + hh * hd;
  for (int d = 0; d < hd; ++d) acc += qp[d] * kp[d];
  s[idx] = acc * 0.125f;   // 1/sqrt(64)
}

// ------------- cross-attention context: ctx[b,e] = sum_t a[b,h(e),t] * v[b,t,e] -------------
__global__ __launch_bounds__(256)
void cross_ctx_kernel(const float* __restrict__ s, const float* __restrict__ vb,
                      float* __restrict__ ctx) {
  const int idx = blockIdx.x * blockDim.x + threadIdx.x;   // B*E
  if (idx >= B_ * E_) return;
  const int b = idx / E_;
  const int e = idx - b * E_;
  const int hh = e / (E_ / NH_);
  float acc = 0.0f;
  const float* sp = s + ((long long)b * NH_ + hh) * T_;
  const float* vp = vb + (long long)b * T_ * E_ + e;
  for (int t = 0; t < T_; ++t) acc += sp[t] * vp[(long long)t * E_];
  ctx[idx] = acc;
}

// ------------- final[b,t,e] = proj[b,t,e] + ctx[b,e] -------------
__global__ __launch_bounds__(256)
void add_bcast_kernel(const float* __restrict__ proj, const float* __restrict__ ctx,
                      float* __restrict__ outp) {
  const long long idx = (long long)blockIdx.x * blockDim.x + threadIdx.x;
  if (idx >= (long long)BT_ * E_) return;
  const int b = (int)(idx / ((long long)T_ * E_));
  const int e = (int)(idx % E_);
  outp[idx] = proj[idx] + ctx[b * E_ + e];
}

// =====================================================================
// host-side orchestration
// =====================================================================
static void launch_gemm(hipStream_t st,
                        const float* A, long long lda, long long sAb, long long sAh,
                        const float* Bm, long long ldb, long long sBb, long long sBh, int transB,
                        const float* bias, float* C, long long ldc, long long sCb, long long sCh,
                        int M, int N, int K, int batches, int nh, float alpha, int act) {
  dim3 grid((N + 63) / 64, (M + 127) / 128, batches * nh);
  gemm_wmma_kernel<<<grid, 128, 0, st>>>(A, lda, sAb, sAh, Bm, ldb, sBb, sBh, transB,
                                         bias, C, ldc, sCb, sCh, M, N, K, nh, alpha, act);
}

extern "C" void kernel_launch(void* const* d_in, const int* in_sizes, int n_in,
                              void* d_out, int out_size, void* d_ws, size_t ws_size,
                              hipStream_t stream) {
  (void)in_sizes; (void)n_in; (void)out_size; (void)ws_size;
  const float* x       = (const float*)d_in[0];
  const float* pred_w  = (const float*)d_in[1];
  const float* pred_b  = (const float*)d_in[2];
  const float* t_wqkv  = (const float*)d_in[3];
  const float* t_bqkv  = (const float*)d_in[4];
  const float* t_wo    = (const float*)d_in[5];
  const float* t_bo    = (const float*)d_in[6];
  const float* t_ln1g  = (const float*)d_in[7];
  const float* t_ln1b  = (const float*)d_in[8];
  const float* t_w1    = (const float*)d_in[9];
  const float* t_b1    = (const float*)d_in[10];
  const float* t_w2    = (const float*)d_in[11];
  const float* t_b2    = (const float*)d_in[12];
  const float* t_ln2g  = (const float*)d_in[13];
  const float* t_ln2b  = (const float*)d_in[14];
  const float* sb_w    = (const float*)d_in[15];
  const float* sb_b    = (const float*)d_in[16];
  const float* bn1_g   = (const float*)d_in[17];
  const float* bn1_b   = (const float*)d_in[18];
  const float* bn1_m   = (const float*)d_in[19];
  const float* bn1_v   = (const float*)d_in[20];
  const float* dt_w    = (const float*)d_in[21];
  const float* dt_b    = (const float*)d_in[22];
  const float* cc_w1   = (const float*)d_in[23];
  const float* cc_b1   = (const float*)d_in[24];
  const float* bn2_g   = (const float*)d_in[25];
  const float* bn2_b   = (const float*)d_in[26];
  const float* bn2_m   = (const float*)d_in[27];
  const float* bn2_v   = (const float*)d_in[28];
  const float* cc_w2   = (const float*)d_in[29];
  const float* cc_b2   = (const float*)d_in[30];
  const float* ce      = (const float*)d_in[31];
  const float* kp_w    = (const float*)d_in[32];
  const float* kp_b    = (const float*)d_in[33];
  const float* ca_wqkv = (const float*)d_in[34];
  const float* ca_bqkv = (const float*)d_in[35];
  const float* ca_wo   = (const float*)d_in[36];
  const float* ca_bo   = (const float*)d_in[37];
  const float* fp_w    = (const float*)d_in[38];
  const float* fp_b    = (const float*)d_in[39];
  const float* df_w    = (const float*)d_in[40];
  const float* df_b    = (const float*)d_in[41];
  float* outp = (float*)d_out;

  // ---------- workspace carve-up (floats) ----------
  float* W = (float*)d_ws;
  size_t off = 0;
  auto alloc = [&](size_t n) { float* p = W + off; off += n; return p; };
  float* h      = alloc((size_t)BT_ * D_);        // transformer state
  float* qkv    = alloc((size_t)BT_ * 3 * D_);    // qkv / tmpD / k,v,final reuse
  float* big    = alloc((size_t)BT_ * FF_);       // attn scores (B*NH*T*T) == FF hidden
  float* obuf   = alloc((size_t)BT_ * D_);        // attn out / fp out
  float* shrd   = alloc((size_t)BT_ * P_);        // backbone features
  float* proj   = alloc((size_t)BT_ * E_);
  float* tscore = alloc((size_t)BT_);
  float* agg    = alloc((size_t)B_ * P_);
  float* c1buf  = alloc((size_t)B_ * 256);
  float* cat    = alloc((size_t)B_ * E_);
  float* qb     = alloc((size_t)B_ * E_);
  float* ctxr   = alloc((size_t)B_ * E_);
  float* ctxb   = alloc((size_t)B_ * E_);
  float* csc    = alloc((size_t)B_ * NH_ * T_);
  float* tmpD   = qkv;                            // [BT,D] scratch (aliases qkv)
  float* kb     = qkv;                            // [BT,E]
  float* vb     = qkv + (size_t)BT_ * E_;         // [BT,E]
  float* finalb = qkv + (size_t)2 * BT_ * E_;     // [BT,E]

  const int hd = D_ / NH_;   // 256
  const long long D3 = 3LL * D_;

  // 1) GlobalAdaptiveTSM
  tsm_kernel<<<BT_, 256, 0, stream>>>(x, pred_w, pred_b, h);

  // 2) two post-norm transformer layers
  for (int l = 0; l < L_; ++l) {
    const float* wqkv = t_wqkv + (size_t)l * D_ * 3 * D_;
    const float* bqkv = t_bqkv + (size_t)l * 3 * D_;
    const float* wo   = t_wo   + (size_t)l * D_ * D_;
    const float* bo   = t_bo   + (size_t)l * D_;
    const float* w1   = t_w1   + (size_t)l * D_ * FF_;
    const float* b1   = t_b1   + (size_t)l * FF_;
    const float* w2   = t_w2   + (size_t)l * FF_ * D_;
    const float* b2   = t_b2   + (size_t)l * D_;

    // qkv = h @ wqkv + bqkv
    launch_gemm(stream, h, D_, 0, 0, wqkv, D3, 0, 0, 0, bqkv,
                qkv, D3, 0, 0, BT_, 3 * D_, D_, 1, 1, 1.0f, 0);
    // scores[b,h] = Q @ K^T / sqrt(hd)
    launch_gemm(stream, qkv, D3, (long long)T_ * D3, hd,
                qkv + D_, D3, (long long)T_ * D3, hd, 1, nullptr,
                big, T_, (long long)NH_ * T_ * T_, (long long)T_ * T_,
                T_, T_, hd, B_, NH_, 1.0f / 16.0f, 0);
    softmax_rows_kernel<<<B_ * NH_ * T_, 256, 0, stream>>>(big, T_);
    // O[b,h] = A @ V
    launch_gemm(stream, big, T_, (long long)NH_ * T_ * T_, (long long)T_ * T_,
                qkv + 2 * D_, D3, (long long)T_ * D3, hd, 0, nullptr,
                obuf, D_, (long long)T_ * D_, hd,
                T_, hd, T_, B_, NH_, 1.0f, 0);
    // a = O @ wo + bo   (into tmpD, aliases qkv which is now free)
    launch_gemm(stream, obuf, D_, 0, 0, wo, D_, 0, 0, 0, bo,
                tmpD, D_, 0, 0, BT_, D_, D_, 1, 1, 1.0f, 0);
    ln_res_kernel<<<BT_, 256, 0, stream>>>(h, tmpD,
                t_ln1g + (size_t)l * D_, t_ln1b + (size_t)l * D_);
    // FF
    launch_gemm(stream, h, D_, 0, 0, w1, FF_, 0, 0, 0, b1,
                big, FF_, 0, 0, BT_, FF_, D_, 1, 1, 1.0f, 1);
    launch_gemm(stream, big, FF_, 0, 0, w2, D_, 0, 0, 0, b2,
                tmpD, D_, 0, 0, BT_, D_, FF_, 1, 1, 1.0f, 0);
    ln_res_kernel<<<BT_, 256, 0, stream>>>(h, tmpD,
                t_ln2g + (size_t)l * D_, t_ln2b + (size_t)l * D_);
  }

  // 3) shared backbone
  launch_gemm(stream, h, D_, 0, 0, sb_w, P_, 0, 0, 0, sb_b,
              shrd, P_, 0, 0, BT_, P_, D_, 1, 1, 1.0f, 0);
  bn_relu_kernel<<<((long long)BT_ * P_ + 255) / 256, 256, 0, stream>>>(
      shrd, bn1_g, bn1_b, bn1_m, bn1_v, (long long)BT_ * P_, P_);

  // 4-5) top-k pooling
  rowdot_kernel<<<BT_, 256, 0, stream>>>(shrd, dt_w, dt_b, tscore, P_);
  topk_agg_kernel<<<B_, 256, 0, stream>>>(shrd, tscore, agg);

  // 6-7) classifier -> logits directly into d_out tail
  launch_gemm(stream, agg, P_, 0, 0, cc_w1, 256, 0, 0, 0, cc_b1,
              c1buf, 256, 0, 0, B_, 256, P_, 1, 1, 1.0f, 0);
  bn_relu_kernel<<<(B_ * 256 + 255) / 256, 256, 0, stream>>>(
      c1buf, bn2_g, bn2_b, bn2_m, bn2_v, (long long)B_ * 256, 256);
  float* logits = outp + (size_t)BT_;
  launch_gemm(stream, c1buf, 256, 0, 0, cc_w2, K_, 0, 0, 0, cc_b2,
              logits, K_, 0, 0, B_, K_, 256, 1, 1, 1.0f, 0);

  // 8) category embedding
  catemb_kernel<<<B_, 256, 0, stream>>>(logits, ce, cat);

  // 9) key projection
  launch_gemm(stream, shrd, P_, 0, 0, kp_w, E_, 0, 0, 0, kp_b,
              proj, E_, 0, 0, BT_, E_, P_, 1, 1, 1.0f, 0);

  // 10-12) cross-attention q/k/v (strided slices of ca_wqkv)
  launch_gemm(stream, cat, E_, 0, 0, ca_wqkv, 3LL * E_, 0, 0, 0, ca_bqkv,
              qb, E_, 0, 0, B_, E_, E_, 1, 1, 1.0f, 0);
  launch_gemm(stream, proj, E_, 0, 0, ca_wqkv + E_, 3LL * E_, 0, 0, 0, ca_bqkv + E_,
              kb, E_, 0, 0, BT_, E_, E_, 1, 1, 1.0f, 0);
  launch_gemm(stream, proj, E_, 0, 0, ca_wqkv + 2 * E_, 3LL * E_, 0, 0, 0, ca_bqkv + 2 * E_,
              vb, E_, 0, 0, BT_, E_, E_, 1, 1, 1.0f, 0);

  // 13-15) cross-attention (Tq = 1)
  cross_scores_kernel<<<(B_ * NH_ * T_ + 255) / 256, 256, 0, stream>>>(qb, kb, csc);
  softmax_rows_kernel<<<B_ * NH_, 256, 0, stream>>>(csc, T_);
  cross_ctx_kernel<<<(B_ * E_ + 255) / 256, 256, 0, stream>>>(csc, vb, ctxr);
  launch_gemm(stream, ctxr, E_, 0, 0, ca_wo, E_, 0, 0, 0, ca_bo,
              ctxb, E_, 0, 0, B_, E_, E_, 1, 1, 1.0f, 0);

  // 16) final = proj + ctx
  add_bcast_kernel<<<((long long)BT_ * E_ + 255) / 256, 256, 0, stream>>>(proj, ctxb, finalb);

  // 17-18) final projection + backbone
  launch_gemm(stream, finalb, E_, 0, 0, fp_w, D_, 0, 0, 0, fp_b,
              obuf, D_, 0, 0, BT_, D_, E_, 1, 1, 1.0f, 0);
  launch_gemm(stream, obuf, D_, 0, 0, sb_w, P_, 0, 0, 0, sb_b,
              shrd, P_, 0, 0, BT_, P_, D_, 1, 1, 1.0f, 0);
  bn_relu_kernel<<<((long long)BT_ * P_ + 255) / 256, 256, 0, stream>>>(
      shrd, bn1_g, bn1_b, bn1_m, bn1_v, (long long)BT_ * P_, P_);

  // 19) segmentation head -> d_out head
  rowdot_kernel<<<BT_, 256, 0, stream>>>(shrd, df_w, df_b, outp, P_);
}
